// OPS_506806141446
// MI455X (gfx1250) — matrix operand order repaired and checked by reading, not compiled, not run on hardware
//
#include <hip/hip_runtime.h>
#include <hip/hip_bf16.h>

// ---------------------------------------------------------------------------
// Problem constants (from reference)
// ---------------------------------------------------------------------------
#define BATCH   16
#define NPTS    4096
#define TOTPTS  (BATCH * NPTS)     // 65536
#define C0      256
#define C1      128
#define C2      64
#define C3      16
#define C4      2
#define FPSK    1024
#define KNNK    32
#define BIGF    1e10f
#define BNEPS   1e-5f

typedef __attribute__((ext_vector_type(16))) __bf16 v16bf;
typedef __attribute__((ext_vector_type(8)))  __bf16 v8bf;
typedef __attribute__((ext_vector_type(8)))  float  v8f;
typedef __attribute__((ext_vector_type(4)))  float  v4f;
typedef __attribute__((ext_vector_type(4)))  unsigned int u32x4;
typedef __attribute__((ext_vector_type(8)))  int    i32x8;
typedef __attribute__((ext_vector_type(4)))  int    i32x4;

__device__ __forceinline__ __bf16 f2bf(float f) { return (__bf16)f; }

__device__ __forceinline__ v16bf cat8(v8bf lo, v8bf hi) {
    return __builtin_shufflevector(lo, hi, 0,1,2,3,4,5,6,7,8,9,10,11,12,13,14,15);
}

__device__ __forceinline__ v8f zero8() {
    v8f z;
#pragma unroll
    for (int i = 0; i < 8; ++i) z[i] = 0.0f;
    return z;
}

// ---------------------------------------------------------------------------
// Tensor Data Mover helpers (CDNA5 D# per CDNA5 ISA).
// Arity differs between toolchains: 5 args (ROCm7.2/clang-22), 6 (clang-23).
// ---------------------------------------------------------------------------
__device__ __forceinline__ i32x4 zero4i() {
    i32x4 z;
#pragma unroll
    for (int i = 0; i < 4; ++i) z[i] = 0;
    return z;
}

__device__ __forceinline__ i32x8 zero8i() {
    i32x8 z;
#pragma unroll
    for (int i = 0; i < 8; ++i) z[i] = 0;
    return z;
}

__device__ __forceinline__ void tdm_load(u32x4 g0, i32x8 g1, i32x4 g2, i32x4 g3) {
#if __clang_major__ >= 23
    __builtin_amdgcn_tensor_load_to_lds(g0, g1, g2, g3, zero8i(), 0);
#else
    __builtin_amdgcn_tensor_load_to_lds(g0, g1, g2, g3, 0);
#endif
}
__device__ __forceinline__ void tdm_store(u32x4 g0, i32x8 g1, i32x4 g2, i32x4 g3) {
#if __clang_major__ >= 23
    __builtin_amdgcn_tensor_store_from_lds(g0, g1, g2, g3, zero8i(), 0);
#else
    __builtin_amdgcn_tensor_store_from_lds(g0, g1, g2, g3, 0);
#endif
}

// Group0: count=1 (user), lds_addr, 57b global_addr, type=2; optional gather bits.
__device__ __forceinline__ u32x4 tdm_g0(unsigned ldsByteOff, const void *gptr, bool gather16) {
    unsigned long long ga = (unsigned long long)gptr;
    u32x4 g0;
    g0[0] = 1u | (gather16 ? 0x80000000u : 0u);   // count=1 | gather_mode (16-bit idx)
    g0[1] = ldsByteOff;                            // lds_addr
    g0[2] = (unsigned)(ga & 0xffffffffu);          // global_addr[31:0]
    g0[3] = (unsigned)((ga >> 32) & 0x01ffffffu) | (2u << 30); // addr[56:32] | type=2
    return g0;
}

// Group1 for a simple 2D (rows x width) tensor of 4-byte elements.
// width/tiles must fit 16-bit tile fields.
__device__ __forceinline__ i32x8 tdm_g1_2d(unsigned width, unsigned rows,
                                           unsigned tile0, unsigned tile1) {
    i32x8 g1;
    g1[0] = (int)(2u << 16);                                   // data_size = 4B
    g1[1] = (int)((width & 0xffffu) << 16);                    // tensor_dim0[15:0]
    g1[2] = (int)((width >> 16) | ((rows & 0xffffu) << 16));   // dim0 hi | tensor_dim1 lo
    g1[3] = (int)((rows >> 16) | (tile0 << 16));               // dim1 hi | tile_dim0
    g1[4] = (int)tile1;                                        // tile_dim1 (tile_dim2=0)
    g1[5] = (int)width;                                        // tensor_dim0_stride lo
    g1[6] = (int)((width & 0xffffu) << 16);                    // stride0 hi=0 | stride1 lo
    g1[7] = (int)(width >> 16);                                // stride1 hi
    return g1;
}

// 1D contiguous transfer of n 4-byte elements (n <= 65535).
__device__ __forceinline__ void tdm_copy_1d(unsigned ldsByteOff, const void *gptr,
                                            unsigned n, bool isStore) {
    u32x4 g0 = tdm_g0(ldsByteOff, gptr, false);
    i32x8 g1 = tdm_g1_2d(n, 1, n, 1);
    if (isStore) tdm_store(g0, g1, zero4i(), zero4i());
    else         tdm_load (g0, g1, zero4i(), zero4i());
}

// ---------------------------------------------------------------------------
// Kernel 1: fused 4-layer pointwise MLP via bf16 WMMA, emits scores + mask
// block = 256 threads (8 waves), each wave computes 16 points end-to-end.
// ---------------------------------------------------------------------------
struct MlpParams {
    const float *x;
    const float *w0, *b0, *w1, *b1, *w2, *b2, *w3, *b3;
    const float *g0, *be0, *m0, *v0;
    const float *g1, *be1, *m1, *v1;
    const float *g2, *be2, *m2, *v2;
    float *scoresOut;   // [TOTPTS, 2]
    int   *mask;        // [TOTPTS]
};

#define MLP_LDS_ELEMS (32768 + 8192 + 1024 + 512 + 16384 + 8192 + 2048) // 69120 bf16

__device__ __forceinline__ void bn_relu_store(
    const v8f &acc, int c, int hi, __bf16 *H, int rowStride,
    const float *bias, const float *g, const float *be,
    const float *m, const float *v)
{
    float b  = bias[c];
    float sc = g[c] * rsqrtf(v[c] + BNEPS);
    float sh = be[c] - m[c] * sc;
#pragma unroll
    for (int i = 0; i < 8; ++i) {
        int row = hi * 8 + i;
        float h = fmaxf(acc[i] + b, 0.0f);
        H[row * rowStride + c] = f2bf(h * sc + sh);
    }
}

__global__ void __launch_bounds__(256) mlp_kernel(MlpParams P)
{
    const int tid  = threadIdx.x;
    const int lane = tid & 31;
    const int wave = tid >> 5;
    const int hi   = lane >> 4;
    const int l16  = lane & 15;
    const int rowBase = blockIdx.x * 128 + wave * 16;

    extern __shared__ __bf16 smMlp[];
    __bf16 *W0T = smMlp;            // [128][256]
    __bf16 *W1T = W0T + 32768;      // [64][128]
    __bf16 *W2T = W1T + 8192;       // [16][64]
    __bf16 *W3T = W2T + 1024;       // [16][32] zero-padded
    __bf16 *H1  = W3T + 512;        // 8 x [16][128]
    __bf16 *H2  = H1 + 16384;       // 8 x [16][64]
    __bf16 *H3  = H2 + 8192;        // 8 x [16][16]

    for (int i = tid; i < 32768; i += 256) { int n = i >> 8, k = i & 255; W0T[i] = f2bf(P.w0[k * 128 + n]); }
    for (int i = tid; i <  8192; i += 256) { int n = i >> 7, k = i & 127; W1T[i] = f2bf(P.w1[k *  64 + n]); }
    for (int i = tid; i <  1024; i += 256) { int n = i >> 6, k = i &  63; W2T[i] = f2bf(P.w2[k *  16 + n]); }
    for (int i = tid; i <   512; i += 256) {
        int n = i >> 5, k = i & 31;
        W3T[i] = (k < 16 && n < 2) ? f2bf(P.w3[k * 2 + n]) : (__bf16)0.0f;
    }
    __syncthreads();

    __bf16 *H1w = H1 + wave * 16 * 128;
    __bf16 *H2w = H2 + wave * 16 * 64;
    __bf16 *H3w = H3 + wave * 16 * 16;
    const int r = rowBase + l16;

    // ---------------- layer 0: [16x256] x [256x128] ----------------
    v8f acc0[8];
#pragma unroll
    for (int nt = 0; nt < 8; ++nt) acc0[nt] = zero8();
#pragma unroll
    for (int kc = 0; kc < 8; ++kc) {
        const float *xr = P.x + (size_t)r * C0 + kc * 32 + (hi ? 8 : 0);
        v4f f0 = *(const v4f *)(xr + 0);
        v4f f1 = *(const v4f *)(xr + 4);
        v4f f2 = *(const v4f *)(xr + 16);
        v4f f3 = *(const v4f *)(xr + 20);
        v16bf a;
#pragma unroll
        for (int j = 0; j < 4; ++j) {
            a[j]      = f2bf(f0[j]);
            a[4 + j]  = f2bf(f1[j]);
            a[8 + j]  = f2bf(f2[j]);
            a[12 + j] = f2bf(f3[j]);
        }
#pragma unroll
        for (int nt = 0; nt < 8; ++nt) {
            const __bf16 *bp = W0T + (nt * 16 + l16) * 256 + kc * 32 + hi * 8;
            v16bf b = cat8(*(const v8bf *)bp, *(const v8bf *)(bp + 16));
            acc0[nt] = __builtin_amdgcn_wmma_f32_16x16x32_bf16(
                false, a, false, b, (short)0, acc0[nt], false, false);
        }
    }
#pragma unroll
    for (int nt = 0; nt < 8; ++nt)
        bn_relu_store(acc0[nt], nt * 16 + l16, hi, H1w, 128, P.b0, P.g0, P.be0, P.m0, P.v0);

    // ---------------- layer 1: [16x128] x [128x64] ----------------
    v8f acc1[4];
#pragma unroll
    for (int nt = 0; nt < 4; ++nt) acc1[nt] = zero8();
#pragma unroll
    for (int kc = 0; kc < 4; ++kc) {
        const __bf16 *ap = H1w + l16 * 128 + kc * 32 + hi * 8;
        v16bf a = cat8(*(const v8bf *)ap, *(const v8bf *)(ap + 16));
#pragma unroll
        for (int nt = 0; nt < 4; ++nt) {
            const __bf16 *bp = W1T + (nt * 16 + l16) * 128 + kc * 32 + hi * 8;
            v16bf b = cat8(*(const v8bf *)bp, *(const v8bf *)(bp + 16));
            acc1[nt] = __builtin_amdgcn_wmma_f32_16x16x32_bf16(
                false, a, false, b, (short)0, acc1[nt], false, false);
        }
    }
#pragma unroll
    for (int nt = 0; nt < 4; ++nt)
        bn_relu_store(acc1[nt], nt * 16 + l16, hi, H2w, 64, P.b1, P.g1, P.be1, P.m1, P.v1);

    // ---------------- layer 2: [16x64] x [64x16] ----------------
    v8f acc2 = zero8();
#pragma unroll
    for (int kc = 0; kc < 2; ++kc) {
        const __bf16 *ap = H2w + l16 * 64 + kc * 32 + hi * 8;
        v16bf a = cat8(*(const v8bf *)ap, *(const v8bf *)(ap + 16));
        const __bf16 *bp = W2T + l16 * 64 + kc * 32 + hi * 8;
        v16bf b = cat8(*(const v8bf *)bp, *(const v8bf *)(bp + 16));
        acc2 = __builtin_amdgcn_wmma_f32_16x16x32_bf16(
            false, a, false, b, (short)0, acc2, false, false);
    }
    bn_relu_store(acc2, l16, hi, H3w, 16, P.b2, P.g2, P.be2, P.m2, P.v2);

    // ---------------- layer 3: [16x16(pad32)] x [16x2(padded)] ----------------
    {
        const __bf16 *ap = H3w + l16 * 16 + hi * 8;
        v16bf a;
#pragma unroll
        for (int j = 0; j < 8; ++j) { a[j] = ap[j]; a[8 + j] = (__bf16)0.0f; }
        const __bf16 *bp = W3T + l16 * 32 + hi * 8;
        v16bf b = cat8(*(const v8bf *)bp, *(const v8bf *)(bp + 16));
        v8f acc3 = zero8();
        acc3 = __builtin_amdgcn_wmma_f32_16x16x32_bf16(
            false, a, false, b, (short)0, acc3, false, false);

        const int c = l16;
        const float bias = (c < 2) ? P.b3[c] : 0.0f;
#pragma unroll
        for (int i = 0; i < 8; ++i) {
            int pt  = rowBase + hi * 8 + i;
            float s = acc3[i] + bias;
            float so = __shfl_xor(s, 1, 32);
            if (c == 0) {
                P.scoresOut[(size_t)pt * 2 + 0] = s;
                P.mask[pt] = (so > s) ? 1 : 0;   // argmax > 0, ties -> class 0
            } else if (c == 1) {
                P.scoresOut[(size_t)pt * 2 + 1] = s;
            }
        }
    }
}

// ---------------------------------------------------------------------------
// Kernel 2: masked FPS, one 1024-thread block per batch.
// pos tile (48KB) DMA'd to LDS via TENSOR_LOAD_TO_LDS (wave 0), overlapped
// with mask loads; 1024 sequential argmax steps with wave32 shfl reductions.
// LDS: pq[4096*3] (interleaved xyz, at offset 0 for TDM) | dist | msk | ...
// ---------------------------------------------------------------------------
#define FPS_LDS_WORDS (NPTS * 3 + NPTS + NPTS + FPSK + 32 + 32 + 4)
#define FPS_LDS_BYTES (FPS_LDS_WORDS * 4)

__global__ void __launch_bounds__(1024) fps_kernel(
    const float *pos, const int *mask, int *idxOut, float *posNewOut)
{
    const int b = blockIdx.x;
    const int tid = threadIdx.x;

    extern __shared__ float smFps[];
    float *pq   = smFps;                 // [4096][3] at LDS offset 0
    float *dist = pq + NPTS * 3;
    int   *msk  = (int *)(dist + NPTS);
    int   *lidx = msk + NPTS;
    float *redv = (float *)(lidx + FPSK);
    int   *redi = (int *)(redv + 32);
    int   *bc   = redi + 32;

    // wave 0: DMA the whole pos tile (4096*3 f32) into LDS
    if (tid < 32) {
        tdm_copy_1d(0u, pos + (size_t)b * NPTS * 3, NPTS * 3, false);
    }
    for (int i = tid; i < NPTS; i += 1024) {
        int m = mask[b * NPTS + i];
        msk[i]  = m;
        dist[i] = m ? BIGF : -BIGF;
    }
    if (tid < 32) __builtin_amdgcn_s_wait_tensorcnt(0);
    __syncthreads();

    // first = index of first masked point
    int lf = 0x7fffffff;
#pragma unroll
    for (int e = 0; e < 4; ++e) {
        int i = tid + e * 1024;
        if (msk[i] && i < lf) lf = i;
    }
#pragma unroll
    for (int o = 16; o > 0; o >>= 1) { int of = __shfl_xor(lf, o, 32); lf = min(lf, of); }
    if ((tid & 31) == 0) redi[tid >> 5] = lf;
    __syncthreads();
    if (tid < 32) {
        int v = redi[tid];
#pragma unroll
        for (int o = 16; o > 0; o >>= 1) { int ov = __shfl_xor(v, o, 32); v = min(v, ov); }
        if (tid == 0) bc[0] = (v == 0x7fffffff) ? 0 : v;
    }
    __syncthreads();
    int last = bc[0];

    for (int t = 0; t < FPSK; ++t) {
        if (tid == 0) { lidx[t] = last; idxOut[b * FPSK + t] = last; }
        float lx = pq[last * 3 + 0], ly = pq[last * 3 + 1], lz = pq[last * 3 + 2];

        float bv = -3.0e38f;
        int   bi = 0x7fffffff;
#pragma unroll
        for (int e = 0; e < 4; ++e) {
            int i = tid + e * 1024;
            if (msk[i]) {
                float dx = pq[i * 3 + 0] - lx;
                float dy = pq[i * 3 + 1] - ly;
                float dz = pq[i * 3 + 2] - lz;
                float d  = dx * dx + dy * dy + dz * dz;
                float nd = fminf(dist[i], d);
                dist[i]  = nd;
                if (nd > bv || (nd == bv && i < bi)) { bv = nd; bi = i; }
            }
        }
#pragma unroll
        for (int o = 16; o > 0; o >>= 1) {
            float ov = __shfl_xor(bv, o, 32);
            int   oi = __shfl_xor(bi, o, 32);
            if (ov > bv || (ov == bv && oi < bi)) { bv = ov; bi = oi; }
        }
        if ((tid & 31) == 0) { redv[tid >> 5] = bv; redi[tid >> 5] = bi; }
        __syncthreads();
        if (tid < 32) {
            float v = redv[tid];
            int   ix = redi[tid];
#pragma unroll
            for (int o = 16; o > 0; o >>= 1) {
                float ov = __shfl_xor(v, o, 32);
                int   oi = __shfl_xor(ix, o, 32);
                if (ov > v || (ov == v && oi < ix)) { v = ov; ix = oi; }
            }
            if (tid == 0) bc[0] = (ix == 0x7fffffff) ? 0 : ix;
        }
        __syncthreads();
        last = bc[0];
    }
    __syncthreads();

    {
        int id = lidx[tid];
        float *o = posNewOut + ((size_t)b * FPSK + tid) * 3;
        o[0] = pq[id * 3 + 0];
        o[1] = pq[id * 3 + 1];
        o[2] = pq[id * 3 + 2];
    }
}

// ---------------------------------------------------------------------------
// Kernel 3: feature gather via TDM gather-mode (16-bit row indices).
// One wave per block; block gathers 16 scattered rows of x and x_origin into
// LDS, then TENSOR_STORE_FROM_LDS writes the 16 contiguous output rows.
// grid = 1024 blocks x 32 threads; LDS = 2 * 16*256 f32 = 32KB.
// ---------------------------------------------------------------------------
__global__ void __launch_bounds__(32) gather_tdm_kernel(
    const float *x, const float *xo, const int *idx, float *xNew, float *xCopy)
{
    const int wg = blockIdx.x;          // each handles 16 consecutive output rows
    const int b  = wg >> 6;             // 64 blocks per batch
    const int rowBase = wg * 16;

    extern __shared__ float smG[];      // LDS offset 0: x rows; +16KB: xo rows
    (void)smG;

    // 16 uniform row indices, packed 16-bit for D# groups 2/3
    int id[16];
#pragma unroll
    for (int j = 0; j < 16; ++j) id[j] = idx[rowBase + j];
    i32x4 gi0, gi1;
#pragma unroll
    for (int j = 0; j < 4; ++j) {
        gi0[j] = (id[2 * j]     & 0xffff) | (id[2 * j + 1] << 16);
        gi1[j] = (id[2 * j + 8] & 0xffff) | (id[2 * j + 9] << 16);
    }

    const size_t srcBase = (size_t)b * NPTS * C0;
    // gather loads: tensor = [4096 rows][256], tile = 16 indexed rows of 256
    {
        u32x4 g0 = tdm_g0(0u, x + srcBase, true);
        i32x8 g1 = tdm_g1_2d(C0, NPTS, C0, 16);   // tile_dim1 = #valid indices
        tdm_load(g0, g1, gi0, gi1);
    }
    {
        u32x4 g0 = tdm_g0(16384u, xo + srcBase, true);
        i32x8 g1 = tdm_g1_2d(C0, NPTS, C0, 16);
        tdm_load(g0, g1, gi0, gi1);
    }
    __builtin_amdgcn_s_wait_tensorcnt(0);

    // contiguous stores: 16 rows x 256 = 4096 elements each
    tdm_copy_1d(0u,     xNew  + (size_t)rowBase * C0, 16 * C0, true);
    tdm_copy_1d(16384u, xCopy + (size_t)rowBase * C0, 16 * C0, true);
    __builtin_amdgcn_s_wait_tensorcnt(0);
}

// ---------------------------------------------------------------------------
// Kernel 4: kNN (k=32, self included) among the 1024 FPS points per batch.
// pos_new tile DMA'd to LDS via TDM; per-thread stable insertion (matches
// jax.lax.top_k tie-breaking). grid (4, BATCH), block 256.
// ---------------------------------------------------------------------------
#define KNN_LDS_BYTES (FPSK * 3 * 4)

__global__ void __launch_bounds__(256) knn_kernel(const float *posNew, float *knnOut)
{
    const int b   = blockIdx.y;
    const int dst = blockIdx.x * 256 + threadIdx.x;

    extern __shared__ float smKnn[];
    float *q = smKnn;                   // [1024][3] at LDS offset 0

    if (threadIdx.x < 32) {
        tdm_copy_1d(0u, posNew + (size_t)b * FPSK * 3, FPSK * 3, false);
        __builtin_amdgcn_s_wait_tensorcnt(0);
    }
    __syncthreads();

    const float ox = q[dst * 3 + 0], oy = q[dst * 3 + 1], oz = q[dst * 3 + 2];
    float bd[KNNK];
    int   bi[KNNK];
#pragma unroll
    for (int j = 0; j < KNNK; ++j) { bd[j] = 3.4e38f; bi[j] = 0x7fffffff; }

    for (int c = 0; c < FPSK; ++c) {
        float dx = q[c * 3 + 0] - ox;
        float dy = q[c * 3 + 1] - oy;
        float dz = q[c * 3 + 2] - oz;
        float d  = dx * dx + dy * dy + dz * dz;
        int   ci = c;
        if (d < bd[KNNK - 1] || (d == bd[KNNK - 1] && ci < bi[KNNK - 1])) {
#pragma unroll
            for (int j = 0; j < KNNK; ++j) {
                bool lt = (d < bd[j]) || (d == bd[j] && ci < bi[j]);
                if (lt) {
                    float td = bd[j]; int ti = bi[j];
                    bd[j] = d; bi[j] = ci;
                    d = td; ci = ti;
                }
            }
        }
    }

    float *o = knnOut + ((size_t)(b * FPSK + dst)) * KNNK;
#pragma unroll
    for (int j = 0; j < KNNK; ++j) o[j] = (float)bi[j];
}

// ---------------------------------------------------------------------------
// Host launcher
// Inputs (setup_inputs dict order, params flattened in insertion order):
//   pos, x, x_origin, w0,b0,w1,b1,w2,b2,w3,b3,
//   bn0_g,bn0_b,bn0_m,bn0_v, bn1_*, bn2_*, batch_size
// Output: x_new | x_copy_new | pos_new | knn_src(float) | scores
// ---------------------------------------------------------------------------
extern "C" void kernel_launch(void* const* d_in, const int* in_sizes, int n_in,
                              void* d_out, int out_size, void* d_ws, size_t ws_size,
                              hipStream_t stream)
{
    const float *pos = (const float *)d_in[0];
    const float *x   = (const float *)d_in[1];
    const float *xo  = (const float *)d_in[2];

    MlpParams P;
    P.x  = x;
    P.w0 = (const float *)d_in[3];  P.b0 = (const float *)d_in[4];
    P.w1 = (const float *)d_in[5];  P.b1 = (const float *)d_in[6];
    P.w2 = (const float *)d_in[7];  P.b2 = (const float *)d_in[8];
    P.w3 = (const float *)d_in[9];  P.b3 = (const float *)d_in[10];
    P.g0 = (const float *)d_in[11]; P.be0 = (const float *)d_in[12];
    P.m0 = (const float *)d_in[13]; P.v0  = (const float *)d_in[14];
    P.g1 = (const float *)d_in[15]; P.be1 = (const float *)d_in[16];
    P.m1 = (const float *)d_in[17]; P.v1  = (const float *)d_in[18];
    P.g2 = (const float *)d_in[19]; P.be2 = (const float *)d_in[20];
    P.m2 = (const float *)d_in[21]; P.v2  = (const float *)d_in[22];

    float *out    = (float *)d_out;
    float *xNew   = out;                                    // 4194304
    float *xCopy  = out + (size_t)BATCH * FPSK * C0;        // 4194304
    float *posNew = xCopy + (size_t)BATCH * FPSK * C0;      // 49152
    float *knn    = posNew + (size_t)BATCH * FPSK * 3;      // 524288
    float *scores = knn + (size_t)BATCH * FPSK * KNNK;      // 131072

    int *mask = (int *)d_ws;            // TOTPTS ints
    int *idx  = mask + TOTPTS;          // BATCH*FPSK ints

    P.scoresOut = scores;
    P.mask      = mask;

    // 1) fused WMMA MLP -> scores + mask
    mlp_kernel<<<TOTPTS / 128, 256, MLP_LDS_ELEMS * sizeof(__bf16), stream>>>(P);

    // 2) masked FPS per batch (TDM-staged pos) -> idx + pos_new
    fps_kernel<<<BATCH, 1024, FPS_LDS_BYTES, stream>>>(pos, mask, idx, posNew);

    // 3) feature gathers: TDM gather-mode load + contiguous TDM store
    gather_tdm_kernel<<<BATCH * FPSK / 16, 32, 2 * 16 * C0 * sizeof(float), stream>>>(
        x, xo, idx, xNew, xCopy);

    // 4) kNN graph on FPS points (TDM-staged pos_new)
    knn_kernel<<<dim3(FPSK / 256, BATCH), 256, KNN_LDS_BYTES, stream>>>(posNew, knn);
}
